// MultiHeadAttention_53584011985455
// MI455X (gfx1250) — matrix operand order repaired and checked by reading, not compiled, not run on hardware
//
#include <hip/hip_runtime.h>
#include <hip/hip_bf16.h>
#include <math.h>

// ---------------------------------------------------------------------------
// MI455X (gfx1250) implementation.
//   - All GEMMs run through v_wmma_f32_16x16x32_bf16 (f32 accumulate).
//   - Weights/activations converted to bf16 once; LN/softmax stay f32.
//   - One generic batched WMMA GEMM, templated on which operand dimension is
//     unit-stride so staging is vectorized (global_load_b128) and coalesced,
//     with register-buffered software pipelining of the next K tile.
// ---------------------------------------------------------------------------

typedef __bf16 bf16_t;
typedef __attribute__((ext_vector_type(16))) __bf16 v16bf;
typedef __attribute__((ext_vector_type(8)))  __bf16 v8bf;
typedef __attribute__((ext_vector_type(8)))  float   v8f;

#define LN_EPS 1e-5f
#define GPAD 72   // LDS row pitch (bf16 elems) for BK=64: 144B, 16B-aligned

// ------------------------------ elementwise --------------------------------

__global__ void cvt_f32_bf16(const float* __restrict__ in, bf16_t* __restrict__ out,
                             long long n) {
  long long i = (long long)blockIdx.x * blockDim.x + threadIdx.x;
  long long stride = (long long)gridDim.x * blockDim.x;
  for (; i < n; i += stride) out[i] = (bf16_t)in[i];
}

// LayerNorm over C=64 per pixel; writes xf[B,HW,64] (bf16) and the pixel-folded
// kv[B,1024,256] (bf16): kv[b, p>>2, c*4 + (p&3)] = xn[b, p, c].
__global__ void ln1_scatter(const float* __restrict__ x,
                            const float* __restrict__ w, const float* __restrict__ bia,
                            bf16_t* __restrict__ xf, bf16_t* __restrict__ kv) {
  const int gw   = (blockIdx.x * blockDim.x + threadIdx.x) >> 5;
  const int lane = threadIdx.x & 31;
  const int bb = gw >> 12;      // / 4096 pixels
  const int p  = gw & 4095;
  const float* xp = x + (long long)bb * 64 * 4096 + p;
  float v0 = xp[(long long)lane * 4096];
  float v1 = xp[(long long)(lane + 32) * 4096];
  float s = v0 + v1;
#pragma unroll
  for (int m = 16; m >= 1; m >>= 1) s += __shfl_xor(s, m, 32);
  const float mean = s * (1.f / 64.f);
  const float d0 = v0 - mean, d1 = v1 - mean;
  float q = d0 * d0 + d1 * d1;
#pragma unroll
  for (int m = 16; m >= 1; m >>= 1) q += __shfl_xor(q, m, 32);
  const float inv = rsqrtf(q * (1.f / 64.f) + LN_EPS);
  const float y0 = d0 * inv * w[lane]      + bia[lane];
  const float y1 = d1 * inv * w[lane + 32] + bia[lane + 32];
  bf16_t* xfp = xf + ((long long)bb * 4096 + p) * 64;
  xfp[lane]      = (bf16_t)y0;
  xfp[lane + 32] = (bf16_t)y1;
  bf16_t* kvp = kv + (long long)bb * 1024 * 256 + (long long)(p >> 2) * 256 + (p & 3);
  kvp[(long long)lane * 4]        = (bf16_t)y0;
  kvp[(long long)(lane + 32) * 4] = (bf16_t)y1;
}

// LayerNorm over rows of 64 f32 -> bf16 out (same layout). One wave per row.
__global__ void ln64_bf16(const float* __restrict__ in,
                          const float* __restrict__ w, const float* __restrict__ bia,
                          bf16_t* __restrict__ out) {
  const int row  = (blockIdx.x * blockDim.x + threadIdx.x) >> 5;
  const int lane = threadIdx.x & 31;
  const float* ip = in + (long long)row * 64;
  float v0 = ip[lane], v1 = ip[lane + 32];
  float s = v0 + v1;
#pragma unroll
  for (int m = 16; m >= 1; m >>= 1) s += __shfl_xor(s, m, 32);
  const float mean = s * (1.f / 64.f);
  const float d0 = v0 - mean, d1 = v1 - mean;
  float q = d0 * d0 + d1 * d1;
#pragma unroll
  for (int m = 16; m >= 1; m >>= 1) q += __shfl_xor(q, m, 32);
  const float inv = rsqrtf(q * (1.f / 64.f) + LN_EPS);
  bf16_t* op = out + (long long)row * 64;
  op[lane]      = (bf16_t)(d0 * inv * w[lane]      + bia[lane]);
  op[lane + 32] = (bf16_t)(d1 * inv * w[lane + 32] + bia[lane + 32]);
}

// Softmax over rows of 128 f32 -> bf16. One wave per row, 4 elems/lane.
__global__ void softmax128_bf16(const float* __restrict__ in, bf16_t* __restrict__ out) {
  const int row  = (blockIdx.x * blockDim.x + threadIdx.x) >> 5;
  const int lane = threadIdx.x & 31;
  const float* ip = in + (long long)row * 128 + lane * 4;
  float a0 = ip[0], a1 = ip[1], a2 = ip[2], a3 = ip[3];
  float mx = fmaxf(fmaxf(a0, a1), fmaxf(a2, a3));
#pragma unroll
  for (int m = 16; m >= 1; m >>= 1) mx = fmaxf(mx, __shfl_xor(mx, m, 32));
  const float e0 = __expf(a0 - mx), e1 = __expf(a1 - mx);
  const float e2 = __expf(a2 - mx), e3 = __expf(a3 - mx);
  float s = e0 + e1 + e2 + e3;
#pragma unroll
  for (int m = 16; m >= 1; m >>= 1) s += __shfl_xor(s, m, 32);
  const float inv = 1.f / s;
  bf16_t* op = out + (long long)row * 128 + lane * 4;
  op[0] = (bf16_t)(e0 * inv); op[1] = (bf16_t)(e1 * inv);
  op[2] = (bf16_t)(e2 * inv); op[3] = (bf16_t)(e3 * inv);
}

// ------------------------------ WMMA GEMM ----------------------------------
// D[z,m,n] = scale * sum_k A[z,m,k] * B[z,k,n] (+ resid).
// Block = 128 threads (4 waves), tile 64(M) x 64(N), BK = 64.
// Wave w owns rows [16w,16w+16), all 64 columns (4 f32 accumulators).
// Requires M%64==0, N%64==0, K%64==0 (true for every GEMM here).
//
// A_KC: A's K dimension is unit stride (As_k==1).  Otherwise As_m==1.
// B_NC: B's N dimension is unit stride (Bs_n==1).  Otherwise Bs_k==1.
// The unit-stride side is loaded as v8bf (global_load_b128), coalesced.
//
// Fragment<->lane maps follow the CDNA5 ISA tables (CDNA5 ISA):
//   A 16x32 bf16 : lane(0-15)=row M, K-halves {kb..kb+7, 16+kb..+7}, kb=8*(lane>>4)
//   B 32x16 bf16 : lane(0-15)=col N, K-halves {kb..kb+7, 16+kb..+7}, kb=8*(lane>>4) (same as A)
//   C/D 16x16 f32: lane = n + 16*(m>=8), vgpr v -> m = 8*(lane>>4)+v
template<bool A_KC, bool B_NC, bool OUT_BF16, bool RESID>
__global__ __launch_bounds__(128)
void gemm_wmma_bf16(const bf16_t* __restrict__ Ag, const bf16_t* __restrict__ Bg,
                    void* __restrict__ Dg, const float* __restrict__ Rg,
                    int M, int N, int K,
                    long long As_m, long long As_k, long long Ab,
                    long long Bs_k, long long Bs_n, long long Bb,
                    long long Ds_m, long long Ds_n, long long Db,
                    float scale) {
  __shared__ __align__(16) bf16_t sA [64 * GPAD];   // [m][k]
  __shared__ __align__(16) bf16_t sBT[64 * GPAD];   // [n][k]  (B transposed)
  const int t    = threadIdx.x;
  const int wave = t >> 5;
  const int lane = t & 31;
  const int m0 = blockIdx.y << 6;
  const int n0 = blockIdx.x << 6;
  const long long z = blockIdx.z;
  Ag += z * Ab;
  Bg += z * Bb;

  v8f acc[4];
#pragma unroll
  for (int j = 0; j < 4; ++j) acc[j] = 0.f;

  // Per-thread staging coordinates (64x64 tile = 512 v8bf chunks, 4/thread).
  const int cRow = t >> 3;          // 0..15 base row (of 64, step 16 per i)
  const int cOff = (t & 7) << 3;    // 0,8,..,56: unit-stride offset within row

  v8bf aR[4], bR[4];

  auto load_tiles = [&](int k0) {
#pragma unroll
    for (int i = 0; i < 4; ++i) {
      const int row = cRow + (i << 4);          // 0..63
      if (A_KC) {   // row = m, cOff along k
        aR[i] = *(const v8bf*)&Ag[(long long)(m0 + row) * As_m + (k0 + cOff)];
      } else {      // row = k, cOff along m (As_m == 1)
        aR[i] = *(const v8bf*)&Ag[(long long)(k0 + row) * As_k + (m0 + cOff)];
      }
      if (B_NC) {   // row = k, cOff along n (Bs_n == 1)
        bR[i] = *(const v8bf*)&Bg[(long long)(k0 + row) * Bs_k + (n0 + cOff)];
      } else {      // row = n, cOff along k (Bs_k == 1)
        bR[i] = *(const v8bf*)&Bg[(long long)(n0 + row) * Bs_n + (k0 + cOff)];
      }
    }
  };

  auto store_tiles = [&]() {
#pragma unroll
    for (int i = 0; i < 4; ++i) {
      const int row = cRow + (i << 4);
      if (A_KC) {
        *(v8bf*)&sA[row * GPAD + cOff] = aR[i];
      } else {      // transpose into sA[m][k]
#pragma unroll
        for (int j = 0; j < 8; ++j) sA[(cOff + j) * GPAD + row] = aR[i][j];
      }
      if (B_NC) {   // transpose into sBT[n][k]
#pragma unroll
        for (int j = 0; j < 8; ++j) sBT[(cOff + j) * GPAD + row] = bR[i][j];
      } else {
        *(v8bf*)&sBT[row * GPAD + cOff] = bR[i];
      }
    }
  };

  const int fm = (wave << 4) + (lane & 15);
  const int kA = (lane >> 4) << 3;    // 0 or 8
  const int kB = (lane >> 4) << 3;    // 0 or 8

  load_tiles(0);
  for (int k0 = 0; k0 < K; k0 += 64) {
    store_tiles();
    __syncthreads();
    if (k0 + 64 < K) load_tiles(k0 + 64);   // overlap next loads with WMMA
#pragma unroll
    for (int kk = 0; kk < 64; kk += 32) {
      const v8bf alo = *(const v8bf*)&sA[fm * GPAD + kk + kA];
      const v8bf ahi = *(const v8bf*)&sA[fm * GPAD + kk + 16 + kA];
      const v16bf afrag = __builtin_shufflevector(alo, ahi,
          0, 1, 2, 3, 4, 5, 6, 7, 8, 9, 10, 11, 12, 13, 14, 15);
#pragma unroll
      for (int j = 0; j < 4; ++j) {
        const int fn = (j << 4) + (lane & 15);
        const v8bf blo = *(const v8bf*)&sBT[fn * GPAD + kk + kB];
        const v8bf bhi = *(const v8bf*)&sBT[fn * GPAD + kk + kB + 16];
        const v16bf bfrag = __builtin_shufflevector(blo, bhi,
            0, 1, 2, 3, 4, 5, 6, 7, 8, 9, 10, 11, 12, 13, 14, 15);
        acc[j] = __builtin_amdgcn_wmma_f32_16x16x32_bf16(
            false, afrag, false, bfrag, (short)0, acc[j], false, false);
      }
    }
    __syncthreads();
  }

  const int dn0 = lane & 15;
  const int dmb = (lane >> 4) << 3;
#pragma unroll
  for (int j = 0; j < 4; ++j) {
    const int n = n0 + (j << 4) + dn0;
#pragma unroll
    for (int v = 0; v < 8; ++v) {
      const int m = m0 + (wave << 4) + dmb + v;
      const long long off = z * Db + (long long)m * Ds_m + (long long)n * Ds_n;
      float val = acc[j][v] * scale;
      if (RESID) val += Rg[off];
      if (OUT_BF16) ((bf16_t*)Dg)[off] = (bf16_t)val;
      else          ((float*) Dg)[off] = val;
    }
  }
}

// ------------------------------- driver ------------------------------------

extern "C" void kernel_launch(void* const* d_in, const int* in_sizes, int n_in,
                              void* d_out, int out_size, void* d_ws, size_t ws_size,
                              hipStream_t stream) {
  (void)in_sizes; (void)n_in; (void)out_size; (void)ws_size;
  const float* x    = (const float*)d_in[0];
  const float* ln1w = (const float*)d_in[1];
  const float* ln1b = (const float*)d_in[2];
  const float* ln2w = (const float*)d_in[3];
  const float* ln2b = (const float*)d_in[4];
  const float* W_rk = (const float*)d_in[5];
  const float* W_rv = (const float*)d_in[6];
  const float* W_q  = (const float*)d_in[7];
  const float* W_k  = (const float*)d_in[8];
  const float* W_v  = (const float*)d_in[9];
  const float* W_o  = (const float*)d_in[10];

  char* ws = (char*)d_ws;
  size_t off = 0;
  auto alloc = [&](size_t bytes) -> void* {
    void* p = ws + off;
    off = (off + bytes + 255) & ~(size_t)255;
    return p;
  };
  bf16_t* xf_bf   = (bf16_t*)alloc(16LL * 4096 * 64 * 2);   // LN1 output [B,HW,C]
  bf16_t* kv_bf   = (bf16_t*)alloc(16LL * 1024 * 256 * 2);  // folded KV input
  bf16_t* wq_bf   = (bf16_t*)alloc(4096LL * 4096 * 2);
  bf16_t* wo_bf   = (bf16_t*)alloc(4096LL * 4096 * 2);
  bf16_t* wk_bf   = (bf16_t*)alloc(1024LL * 1024 * 2);
  bf16_t* wv_bf   = (bf16_t*)alloc(1024LL * 1024 * 2);
  bf16_t* wrk_bf  = (bf16_t*)alloc(256LL * 64 * 2);
  bf16_t* wrv_bf  = (bf16_t*)alloc(256LL * 64 * 2);
  float*  kraw    = (float*) alloc(16LL * 1024 * 64 * 4);   // kv@W_rk (pre-LN)
  float*  vraw    = (float*) alloc(16LL * 1024 * 64 * 4);
  bf16_t* kred_bf = (bf16_t*)alloc(16LL * 1024 * 64 * 2);   // LN2(k) [B,1024,64]
  bf16_t* vred_bf = (bf16_t*)alloc(16LL * 1024 * 64 * 2);
  bf16_t* q_bf    = (bf16_t*)alloc(16LL * 4096 * 64 * 2);   // q^T [B,HW,C], pre-scaled
  bf16_t* kp_bf   = (bf16_t*)alloc(16LL * 1024 * 64 * 2);   // k-proj^T [B,HW/R,C]
  bf16_t* vp_bf   = (bf16_t*)alloc(16LL * 1024 * 64 * 2);   // v-proj^T
  float*  S_f32   = (float*) alloc(16LL * 8 * 512 * 128 * 4);
  bf16_t* attn_bf = (bf16_t*)alloc(16LL * 8 * 512 * 128 * 2);
  bf16_t* ot_bf   = (bf16_t*)alloc(16LL * 4096 * 64 * 2);   // concat-head O^T

  const float qscale = 1.0f / sqrtf(512.0f);  // fold attention scale into q

  // Weight conversions f32 -> bf16 (read once)
  cvt_f32_bf16<<<1024, 256, 0, stream>>>(W_q,  wq_bf,  4096LL * 4096);
  cvt_f32_bf16<<<1024, 256, 0, stream>>>(W_o,  wo_bf,  4096LL * 4096);
  cvt_f32_bf16<<< 256, 256, 0, stream>>>(W_k,  wk_bf,  1024LL * 1024);
  cvt_f32_bf16<<< 256, 256, 0, stream>>>(W_v,  wv_bf,  1024LL * 1024);
  cvt_f32_bf16<<<  16, 256, 0, stream>>>(W_rk, wrk_bf, 256LL * 64);
  cvt_f32_bf16<<<  16, 256, 0, stream>>>(W_rv, wrv_bf, 256LL * 64);

  // LN1 + scatter (65536 pixels, 8 waves/block)
  ln1_scatter<<<8192, 256, 0, stream>>>(x, ln1w, ln1b, xf_bf, kv_bf);

  // kv @ W_rk / W_rv : M=1024 N=64 K=256, batch=16   (A k-contig, B n-contig)
  gemm_wmma_bf16<true, true, false, false><<<dim3(1, 16, 16), 128, 0, stream>>>(
      kv_bf, wrk_bf, (void*)kraw, nullptr, 1024, 64, 256,
      256, 1, 1024LL * 256,   64, 1, 0,   64, 1, 1024LL * 64, 1.0f);
  gemm_wmma_bf16<true, true, false, false><<<dim3(1, 16, 16), 128, 0, stream>>>(
      kv_bf, wrv_bf, (void*)vraw, nullptr, 1024, 64, 256,
      256, 1, 1024LL * 256,   64, 1, 0,   64, 1, 1024LL * 64, 1.0f);

  // LN2 (16384 rows of 64 each)
  ln64_bf16<<<2048, 256, 0, stream>>>(kraw, ln2w, ln2b, kred_bf);
  ln64_bf16<<<2048, 256, 0, stream>>>(vraw, ln2w, ln2b, vred_bf);

  // q = xf^T @ W_q : M=64 N=4096 K=4096; transposed-A view, store [B,HW,C]*scale
  gemm_wmma_bf16<false, true, true, false><<<dim3(64, 1, 16), 128, 0, stream>>>(
      xf_bf, wq_bf, (void*)q_bf, nullptr, 64, 4096, 4096,
      1, 64, 4096LL * 64,   4096, 1, 0,   1, 64, 4096LL * 64, qscale);

  // k = kred^T @ W_k, v = vred^T @ W_v : M=64 N=1024 K=1024; store transposed
  gemm_wmma_bf16<false, true, true, false><<<dim3(16, 1, 16), 128, 0, stream>>>(
      kred_bf, wk_bf, (void*)kp_bf, nullptr, 64, 1024, 1024,
      1, 64, 1024LL * 64,   1024, 1, 0,   1, 64, 1024LL * 64, 1.0f);
  gemm_wmma_bf16<false, true, true, false><<<dim3(16, 1, 16), 128, 0, stream>>>(
      vred_bf, wv_bf, (void*)vp_bf, nullptr, 64, 1024, 1024,
      1, 64, 1024LL * 64,   1024, 1, 0,   1, 64, 1024LL * 64, 1.0f);

  // Scores S = qh @ kh^T : per z=(b,h): M=512 N=128 K=64 (B is k-contig)
  gemm_wmma_bf16<true, false, false, false><<<dim3(2, 8, 128), 128, 0, stream>>>(
      q_bf, kp_bf, (void*)S_f32, nullptr, 512, 128, 64,
      64, 1, 512LL * 64,   1, 64, 128LL * 64,   128, 1, 512LL * 128, 1.0f);

  // Softmax over 65536 rows of 128
  softmax128_bf16<<<8192, 256, 0, stream>>>(S_f32, attn_bf);

  // O = attn @ vh : per z: M=512 N=64 K=128; store row-major [B,HW,C] bf16
  gemm_wmma_bf16<true, true, true, false><<<dim3(1, 8, 128), 128, 0, stream>>>(
      attn_bf, vp_bf, (void*)ot_bf, nullptr, 512, 64, 128,
      128, 1, 512LL * 128,   64, 1, 128LL * 64,   64, 1, 512LL * 64, 1.0f);

  // out = O^T @ W_o + x : M=64 N=4096 K=4096, f32 out, fused residual
  gemm_wmma_bf16<false, true, false, true><<<dim3(64, 1, 16), 128, 0, stream>>>(
      ot_bf, wo_bf, d_out, x, 64, 4096, 4096,
      1, 64, 4096LL * 64,   4096, 1, 0,   4096, 1, 64LL * 4096, 1.0f);
}
